// LongConvModel_65197603553741
// MI455X (gfx1250) — matrix operand order repaired and checked by reading, not compiled, not run on hardware
//
#include <hip/hip_runtime.h>
#include <hip/hip_bf16.h>
#include <math.h>

#define B_ 8
#define S_ 8192
#define H_ 768
#define N_ 16384
#define NS_ 128           // N_ = NS_ * NS_
#define LAM_ 0.1f

typedef __bf16 bf16;
typedef __attribute__((ext_vector_type(16))) __bf16 v16bf;
typedef __attribute__((ext_vector_type(8)))  float  v8f;
typedef __attribute__((ext_vector_type(2)))  float  v2f;
typedef __attribute__((ext_vector_type(4)))  unsigned int u32x4;
typedef __attribute__((ext_vector_type(4)))  int i32x4;
typedef __attribute__((ext_vector_type(8)))  int i32x8;

#if __has_builtin(__builtin_amdgcn_tensor_load_to_lds) && \
    __has_builtin(__builtin_amdgcn_s_wait_tensorcnt)
#define USE_TDM 1
#else
#define USE_TDM 0
#endif

// LDS layout of k_conv (floats): Xr | Xi | Yr | Yi | ctab(128) | stage(8192)
#define CONV_STAGE_FOFF (4 * N_ + 128)
#define CONV_STAGE_BYTEOFF (CONV_STAGE_FOFF * 4)
#define CONV_LDS_BYTES ((4 * N_ + 128 + S_) * 4)

// ---------------------------------------------------------------------------
// Kernel 1: tiled transpose x (B,S,H) -> u (B,H,S), f32, coalesced both sides
// ---------------------------------------------------------------------------
__global__ __launch_bounds__(256) void k_transpose(const float* __restrict__ x,
                                                   float* __restrict__ u) {
    __shared__ float tile[32][33];
    const int h0 = blockIdx.x * 32;
    const int s0 = blockIdx.y * 32;
    const int b  = blockIdx.z;
    const int tx = threadIdx.x, ty = threadIdx.y;
    const float* xb = x + (size_t)b * S_ * H_;
#pragma unroll
    for (int i = 0; i < 4; ++i) {
        int r = ty + i * 8;
        tile[r][tx] = xb[(size_t)(s0 + r) * H_ + h0 + tx];
    }
    __syncthreads();
    float* ub = u + (size_t)b * H_ * S_;
#pragma unroll
    for (int i = 0; i < 4; ++i) {
        int r = ty + i * 8;
        ub[(size_t)(h0 + r) * S_ + s0 + tx] = tile[tx][r];
    }
}

// ---------------------------------------------------------------------------
// Kernel 2: W f32 -> bf16
// ---------------------------------------------------------------------------
__global__ __launch_bounds__(256) void k_cvt(const float* __restrict__ W,
                                             bf16* __restrict__ Wb, int n) {
    int i = blockIdx.x * 256 + threadIdx.x;
    if (i < n) Wb[i] = (bf16)W[i];
}

// ---------------------------------------------------------------------------
// TDM: async DMA of one contiguous 8192-float row of u into LDS staging.
// D# per CDNA5 ISA ch.8: group0 = {count=1, lds_addr, global_addr, type=2},
// group1 = {data_size=4B, tensor_dim0=8192, tensor_dim1=1, tile_dim0=8192,
//           tile_dim1=1, dim0_stride=8192}. Groups 2/3 zero (<=2D tensor).
// ---------------------------------------------------------------------------
#if USE_TDM
__device__ __forceinline__ void tdm_load_row(const float* gsrc) {
    unsigned long long ga = (unsigned long long)(const void*)gsrc;
    u32x4 g0 = { 1u,                                   // count=1, user mode
                 (unsigned)CONV_STAGE_BYTEOFF,         // lds_addr (bytes)
                 (unsigned)ga,                         // global_addr[31:0]
                 ((unsigned)(ga >> 32) & 0x01FFFFFFu) | 0x80000000u }; // addr[56:32]|type=2
    i32x8 g1 = { 0x00020000,   // data_size=2 (4 bytes), no multicast/pad/iterate
                 0x20000000,   // tensor_dim0 low16 -> word1[31:16] = 8192
                 0x00010000,   // tensor_dim0 hi16=0 | tensor_dim1 low16 = 1
                 0x20000000,   // tensor_dim1 hi=0 | tile_dim0 = 8192
                 1,            // tile_dim1 = 1, tile_dim2 = 0
                 8192,         // tensor_dim0_stride[31:0]
                 0, 0 };
    i32x4 z4 = { 0, 0, 0, 0 };
#if defined(__clang_major__) && __clang_major__ >= 23
    i32x8 z8 = { 0, 0, 0, 0, 0, 0, 0, 0 };
    __builtin_amdgcn_tensor_load_to_lds(g0, g1, z4, z4, z8, 0);
#else
    __builtin_amdgcn_tensor_load_to_lds(g0, g1, z4, z4, 0);
#endif
}
#endif

// ---------------------------------------------------------------------------
// 128-point DFT as 128x128x128 complex matmul on V_WMMA_F32_16X16X4_F32.
// Fs[m][k] = cos(2*pi*m*k/128) + i*s*sin(...) ; s = -1 fwd, +1 inv.
// sin/cos from one 128-entry LDS cosine table (sin t = cos(t-32)).
// ---------------------------------------------------------------------------
__device__ __forceinline__ void dft_left(const float* __restrict__ ctab,
                                         const float* __restrict__ Xr,
                                         const float* __restrict__ Xi,
                                         float* __restrict__ Yr,
                                         float* __restrict__ Yi,
                                         int tid, float s) {
    const int lane = tid & 31, wave = tid >> 5;
    const int half = lane >> 4, lm = lane & 15;
#pragma unroll 1
    for (int t = wave * 2; t < wave * 2 + 2; ++t) {
        const int mi = t >> 3, ni = t & 7;
        v8f P = {}, Q = {}, R = {}, T = {};
#pragma unroll 1
        for (int k0 = 0; k0 < 32; ++k0) {
            v2f aFr, aFi, bXr, bXi;
#pragma unroll
            for (int e = 0; e < 2; ++e) {
                int kk = k0 * 4 + 2 * half + e;
                int gm = mi * 16 + lm;
                int tt = (gm * kk) & 127;
                aFr[e] = ctab[tt];
                aFi[e] = ctab[(tt + 96) & 127];   // sin(th) = cos(th - pi/2)
                bXr[e] = Xr[kk * NS_ + ni * 16 + lm];
                bXi[e] = Xi[kk * NS_ + ni * 16 + lm];
            }
            P = __builtin_amdgcn_wmma_f32_16x16x4_f32(false, aFr, false, bXr, (short)0, P, false, false);
            Q = __builtin_amdgcn_wmma_f32_16x16x4_f32(false, aFi, false, bXi, (short)0, Q, false, false);
            R = __builtin_amdgcn_wmma_f32_16x16x4_f32(false, aFr, false, bXi, (short)0, R, false, false);
            T = __builtin_amdgcn_wmma_f32_16x16x4_f32(false, aFi, false, bXr, (short)0, T, false, false);
        }
#pragma unroll
        for (int r = 0; r < 8; ++r) {
            int gm = mi * 16 + r + 8 * half;
            int gn = ni * 16 + lm;
            Yr[gm * NS_ + gn] = P[r] - s * Q[r];
            Yi[gm * NS_ + gn] = R[r] + s * T[r];
        }
    }
}

__device__ __forceinline__ void dft_right(const float* __restrict__ ctab,
                                          const float* __restrict__ Xr,
                                          const float* __restrict__ Xi,
                                          float* __restrict__ Yr,
                                          float* __restrict__ Yi,
                                          int tid, float s) {
    const int lane = tid & 31, wave = tid >> 5;
    const int half = lane >> 4, lm = lane & 15;
#pragma unroll 1
    for (int t = wave * 2; t < wave * 2 + 2; ++t) {
        const int mi = t >> 3, ni = t & 7;
        v8f P = {}, Q = {}, R = {}, T = {};
#pragma unroll 1
        for (int k0 = 0; k0 < 32; ++k0) {
            v2f aXr, aXi, bFr, bFi;
#pragma unroll
            for (int e = 0; e < 2; ++e) {
                int kk = k0 * 4 + 2 * half + e;
                int gn = ni * 16 + lm;
                int tt = (kk * gn) & 127;
                aXr[e] = Xr[(mi * 16 + lm) * NS_ + kk];
                aXi[e] = Xi[(mi * 16 + lm) * NS_ + kk];
                bFr[e] = ctab[tt];
                bFi[e] = ctab[(tt + 96) & 127];
            }
            P = __builtin_amdgcn_wmma_f32_16x16x4_f32(false, aXr, false, bFr, (short)0, P, false, false);
            Q = __builtin_amdgcn_wmma_f32_16x16x4_f32(false, aXi, false, bFi, (short)0, Q, false, false);
            R = __builtin_amdgcn_wmma_f32_16x16x4_f32(false, aXi, false, bFr, (short)0, R, false, false);
            T = __builtin_amdgcn_wmma_f32_16x16x4_f32(false, aXr, false, bFi, (short)0, T, false, false);
        }
#pragma unroll
        for (int r = 0; r < 8; ++r) {
            int gm = mi * 16 + r + 8 * half;
            int gn = ni * 16 + lm;
            Yr[gm * NS_ + gn] = P[r] - s * Q[r];
            Yi[gm * NS_ + gn] = R[r] + s * T[r];
        }
    }
}

__device__ __forceinline__ void twiddle(float* __restrict__ Yr,
                                        float* __restrict__ Yi,
                                        int tid, float sgn) {
    const float w0 = sgn * 6.28318530717958647f / (float)N_;
#pragma unroll
    for (int e = 0; e < 16; ++e) {
        int j  = tid * 16 + e;
        int k1 = j >> 7, n2 = j & 127;
        float sn, cs;
        __sincosf(w0 * (float)(k1 * n2), &sn, &cs);
        float yr = Yr[j], yi = Yi[j];
        Yr[j] = yr * cs - yi * sn;
        Yi[j] = yr * sn + yi * cs;
    }
}

// ---------------------------------------------------------------------------
// Kernel 3: per-h WMMA-FFT convolution (Monarch 128x128, f32 WMMA) with
// TDM double-buffered prefetch of the next batch's u row into LDS staging.
// ---------------------------------------------------------------------------
__global__ __launch_bounds__(1024) void k_conv(const float* __restrict__ kern,
                                               const float* __restrict__ Dv,
                                               const float* __restrict__ u,
                                               bf16* __restrict__ v) {
    extern __shared__ float smem[];
    float* Xr = smem;
    float* Xi = smem + N_;
    float* Yr = smem + 2 * N_;
    float* Yi = smem + 3 * N_;
    float* ctab  = smem + 4 * N_;           // 128 floats
    float* stage = smem + CONV_STAGE_FOFF;  // 8192 floats

    const int h   = blockIdx.x;
    const int tid = threadIdx.x;
    const int NT  = 1024;

#if USE_TDM
    if (tid < 32) tdm_load_row(u + (size_t)h * S_);   // b=0 row, overlaps k-FFT
#endif

    if (tid < 128) ctab[tid] = __cosf(6.28318530717958647f * (float)tid / 128.0f);

    // soft-thresholded kernel row -> X (row-major 128x128 == natural order)
    for (int i = tid; i < S_; i += NT) {
        float kv = kern[(size_t)h * S_ + i];
        float a  = fabsf(kv) - LAM_;
        a = a > 0.f ? a : 0.f;
        Xr[i] = copysignf(a, kv);
        Xi[i] = 0.f;
    }
    for (int i = S_ + tid; i < N_; i += NT) { Xr[i] = 0.f; Xi[i] = 0.f; }
    __syncthreads();

    // forward: X -> Y -> X   (Out[k1][k2] layout)
    dft_left(ctab, Xr, Xi, Yr, Yi, tid, -1.f);
    __syncthreads();
    twiddle(Yr, Yi, tid, -1.f);
    __syncthreads();
    dft_right(ctab, Yr, Yi, Xr, Xi, tid, -1.f);
    __syncthreads();

    float kfr[16], kfi[16];
#pragma unroll
    for (int e = 0; e < 16; ++e) {
        kfr[e] = Xr[tid * 16 + e];
        kfi[e] = Xi[tid * 16 + e];
    }
    const float dh   = Dv[h];
    const float invN = 1.0f / (float)N_;

#pragma unroll 1
    for (int b = 0; b < B_; ++b) {
        const float* ub = u + ((size_t)b * H_ + h) * (size_t)S_;
#if USE_TDM
        if (tid < 32) __builtin_amdgcn_s_wait_tensorcnt(0);
        __syncthreads();                    // stage ready for everyone
        for (int i = tid; i < S_; i += NT) { Xr[i] = stage[i]; Xi[i] = 0.f; }
        for (int i = S_ + tid; i < N_; i += NT) { Xr[i] = 0.f; Xi[i] = 0.f; }
        __syncthreads();                    // stage consumed
        if (b + 1 < B_ && tid < 32)
            tdm_load_row(u + ((size_t)(b + 1) * H_ + h) * (size_t)S_);
#else
        __syncthreads();
        for (int i = tid; i < S_; i += NT) { Xr[i] = ub[i]; Xi[i] = 0.f; }
        for (int i = S_ + tid; i < N_; i += NT) { Xr[i] = 0.f; Xi[i] = 0.f; }
        __syncthreads();
#endif

        // forward transform of u
        dft_left(ctab, Xr, Xi, Yr, Yi, tid, -1.f);
        __syncthreads();
        twiddle(Yr, Yi, tid, -1.f);
        __syncthreads();
        dft_right(ctab, Yr, Yi, Xr, Xi, tid, -1.f);
        __syncthreads();

        // spectral pointwise multiply (consistent [k1][k2] layout) + 1/N
#pragma unroll
        for (int e = 0; e < 16; ++e) {
            int j = tid * 16 + e;
            float ur = Xr[j], ui = Xi[j];
            Xr[j] = (ur * kfr[e] - ui * kfi[e]) * invN;
            Xi[j] = (ur * kfi[e] + ui * kfr[e]) * invN;
        }
        __syncthreads();

        // inverse: right(G), conj twiddle, left(G)
        dft_right(ctab, Xr, Xi, Yr, Yi, tid, 1.f);
        __syncthreads();
        twiddle(Yr, Yi, tid, 1.f);
        __syncthreads();
        dft_left(ctab, Yr, Yi, Xr, Xi, tid, 1.f);
        __syncthreads();

        // v = silu(conv + D[h]*u), bf16, (B,H,L) layout
        bf16* vb = v + ((size_t)b * H_ + h) * (size_t)S_;
        for (int i = tid; i < S_; i += NT) {
            float y  = Xr[i] + dh * ub[i];
            float sg = 1.0f / (1.0f + __expf(-y));
            vb[i] = (bf16)(y * sg);
        }
    }
}

// ---------------------------------------------------------------------------
// Kernel 4: WMMA bf16 GEMM + bias + GLU + residual.
// A tile staged in fragment-ready order: reader does one contiguous 32B
// LDS vector load per k-step instead of 16 scalar gathers.
// ---------------------------------------------------------------------------
__global__ __launch_bounds__(256) void k_gemm(const bf16* __restrict__ v,
                                              const bf16* __restrict__ Wb,
                                              const float* __restrict__ bias,
                                              const float* __restrict__ x,
                                              float* __restrict__ out) {
    __shared__ __align__(32) bf16 Ap[32][32];    // [m][half*16+e] fragment-ready
    __shared__ __align__(32) bf16 Bs[256][32];   // [n][k]
    __shared__ float zbuf[32][256];

    const int nA0 = blockIdx.x * 128;
    const int l0  = blockIdx.y * 32;
    const int b   = blockIdx.z;
    const int tid  = threadIdx.x;
    const int lane = tid & 31;
    const int wave = tid >> 5;
    const int mW = wave >> 2;
    const int nW = wave & 3;
    const int half = lane >> 4;
    const int lm   = lane & 15;

    v8f acc[4];
#pragma unroll
    for (int j = 0; j < 4; ++j) acc[j] = (v8f){0.f,0.f,0.f,0.f,0.f,0.f,0.f,0.f};

    const int c  = tid;
    const int gn = (c < 128) ? (nA0 + c) : (H_ + nA0 + (c - 128));

    for (int k0 = 0; k0 < H_; k0 += 32) {
        {   // A tile: invert k(e) = 16*half + e at store time
            int kk = tid >> 3;
            int mm = (tid & 7) * 4;
            int hw = kk >> 4;
            int ew = kk & 15;
            int col = hw * 16 + ew;
            const bf16* src = v + ((size_t)b * H_ + k0 + kk) * (size_t)S_ + l0 + mm;
#pragma unroll
            for (int q = 0; q < 4; ++q) Ap[mm + q][col] = src[q];
        }
        {
            const uint4* src = (const uint4*)(Wb + (size_t)gn * H_ + k0);
            uint4* dst = (uint4*)&Bs[c][0];
            dst[0] = src[0];
            dst[1] = src[1];
            if (k0 + 32 < H_)
                __builtin_prefetch(Wb + (size_t)gn * H_ + k0 + 32, 0, 1);
        }
        __syncthreads();

        v16bf a = *(const v16bf*)&Ap[mW * 16 + lm][half * 16];
#pragma unroll
        for (int j = 0; j < 4; ++j) {
            v16bf bfrag = *(const v16bf*)&Bs[nW * 64 + j * 16 + lm][half * 16];
            acc[j] = __builtin_amdgcn_wmma_f32_16x16x32_bf16(
                false, a, false, bfrag, (short)0, acc[j], false, false);
        }
        __syncthreads();
    }

#pragma unroll
    for (int j = 0; j < 4; ++j) {
#pragma unroll
        for (int r = 0; r < 8; ++r) {
            zbuf[mW * 16 + r + 8 * half][nW * 64 + j * 16 + lm] = acc[j][r];
        }
    }
    __syncthreads();

    const int col = tid & 127;
    const int rb  = tid >> 7;
    const float ba = bias[nA0 + col];
    const float bg = bias[H_ + nA0 + col];
    const float* xb = x + ((size_t)b * S_ + l0) * H_ + nA0;
    float* ob = out + ((size_t)b * S_ + l0) * H_ + nA0;
#pragma unroll 1
    for (int r = 0; r < 16; ++r) {
        int row = rb * 16 + r;
        float aa = zbuf[row][col] + ba;
        float gg = zbuf[row][128 + col] + bg;
        float sg = 1.0f / (1.0f + __expf(-gg));
        ob[(size_t)row * H_ + col] = aa * sg + xb[(size_t)row * H_ + col];
    }
}

// ---------------------------------------------------------------------------
extern "C" void kernel_launch(void* const* d_in, const int* in_sizes, int n_in,
                              void* d_out, int out_size, void* d_ws, size_t ws_size,
                              hipStream_t stream) {
    const float* x    = (const float*)d_in[0];
    const float* kern = (const float*)d_in[1];
    const float* Dv   = (const float*)d_in[2];
    const float* W    = (const float*)d_in[3];
    const float* bias = (const float*)d_in[4];
    float* out = (float*)d_out;

    char* ws = (char*)d_ws;
    float* u  = (float*)ws;                                         // B*H*S f32 (192 MB)
    bf16*  v  = (bf16*)(ws + (size_t)B_ * H_ * S_ * 4);             // B*H*S bf16 (96 MB)
    bf16*  Wb = (bf16*)(ws + (size_t)B_ * H_ * S_ * 4
                           + (size_t)B_ * H_ * S_ * 2);             // 2H*H bf16

    dim3 gT(H_ / 32, S_ / 32, B_);
    k_transpose<<<gT, dim3(32, 8, 1), 0, stream>>>(x, u);

    int nW = 2 * H_ * H_;
    k_cvt<<<(nW + 255) / 256, 256, 0, stream>>>(W, Wb, nW);

    k_conv<<<dim3(H_, 1, 1), dim3(1024, 1, 1), CONV_LDS_BYTES, stream>>>(
        kern, Dv, u, v);

    dim3 gG(H_ / 128, S_ / 32, B_);
    k_gemm<<<gG, dim3(256, 1, 1), 0, stream>>>(v, Wb, bias, x, out);
}
